// GAT_56985626083712
// MI455X (gfx1250) — matrix operand order repaired and checked by reading, not compiled, not run on hardware
//
#include <hip/hip_runtime.h>
#include <hip/hip_bf16.h>

typedef __attribute__((ext_vector_type(16))) _Float16 v16h;
typedef __attribute__((ext_vector_type(8)))  float    v8f;

#define NEG_SLOPE 0.2f

// ---------------------------------------------------------------------------
// Fused 3-matrix WMMA GEMM:  Cm[M,NC] = A[M,KC] @ Wm[KC,NC] + bm   (m=0,1,2)
//
// Block = 256 threads = 8 waves; each wave owns one 16-row strip of A
// (converted to f16 fragments once, held in registers for all 3 matrices).
// Each weight matrix is staged in LDS as f16, COLUMN-major [n][k], so a
// lane's B fragment (K = q*32 + e + 16*g, fixed col) is one contiguous
// 32-byte chunk -> ds_load_b128 pairs. All tile loops fully unrolled:
// 3 * (NC/16) * (KC/32) v_wmma_f32_16x16x32_f16 per wave.
// ---------------------------------------------------------------------------
template<int KC, int NC>
__global__ __launch_bounds__(256)
void gat_fused3_gemm(const float* __restrict__ A,
                     const float* __restrict__ W0, const float* __restrict__ W1,
                     const float* __restrict__ W2,
                     const float* __restrict__ b0, const float* __restrict__ b1,
                     const float* __restrict__ b2,
                     float* __restrict__ C0, float* __restrict__ C1,
                     float* __restrict__ C2,
                     int M) {
    extern __shared__ char smem[];                 // NC*KC f16 (<= 32 KB)
    _Float16* __restrict__ bsh = (_Float16*)smem;

    const int tid  = threadIdx.x;
    const int lane = tid & 31;
    const int wv   = tid >> 5;                     // wave id 0..7
    const int g    = lane >> 4;                    // lane-half
    const int ln   = lane & 15;
    const int tm   = blockIdx.x * 8 + wv;          // M tile for this wave
    const bool active = (tm * 16) < M;             // wave-uniform

    // ---- A fragments for the whole K range, converted to f16 once ----
    v16h afr[KC / 32];
    if (active) {
        const float* __restrict__ arow = A + (size_t)(tm * 16 + ln) * KC;
#pragma unroll
        for (int q = 0; q < KC / 32; ++q) {
#pragma unroll
            for (int e = 0; e < 16; ++e) {
                const int ka = q * 32 + g * 16 + e;
                afr[q][e] = (_Float16)arow[ka];
            }
        }
    }

#pragma unroll
    for (int m = 0; m < 3; ++m) {
        const float* __restrict__ Wm   = (m == 0) ? W0 : ((m == 1) ? W1 : W2);
        const float* __restrict__ bias = (m == 0) ? b0 : ((m == 1) ? b1 : b2);
        float* __restrict__       Cm   = (m == 0) ? C0 : ((m == 1) ? C1 : C2);

        // stage Wm into LDS, f16 column-major [n][k]
        for (int i = tid; i < NC * KC; i += 256) {
            const int n = i / KC;
            const int k = i - n * KC;
            bsh[i] = (_Float16)Wm[(size_t)k * NC + n];
        }
        __syncthreads();

        if (active) {
#pragma unroll
            for (int tn = 0; tn < NC / 16; ++tn) {
                const int col = tn * 16 + ln;
                const _Float16* __restrict__ bcol = bsh + (size_t)col * KC;
                v8f c = {};
#pragma unroll
                for (int q = 0; q < KC / 32; ++q) {
                    const v16h b = *(const v16h*)(bcol + q * 32 + g * 16);
                    c = __builtin_amdgcn_wmma_f32_16x16x32_f16(
                            false, afr[q], false, b, (short)0, c, false, false);
                }
                const float bc = bias[col];
#pragma unroll
                for (int r = 0; r < 8; ++r)
                    Cm[(size_t)(tm * 16 + r + g * 8) * NC + col] = c[r] + bc;
            }
        }
        __syncthreads();
    }
}

// ---------------------------------------------------------------------------
// Edge attention score: e = dot(leaky_relu(xl[src] + xr[dst]), att)
// One wave per edge; float4 gathers (one b128 per lane covers H=128 exactly);
// wave32 butterfly reduction; encoded-float atomicMax for segment max.
// ---------------------------------------------------------------------------
__device__ __forceinline__ unsigned enc_f32(float v) {
    unsigned b = __float_as_uint(v);
    return (b & 0x80000000u) ? ~b : (b | 0x80000000u);
}
__device__ __forceinline__ float dec_f32(unsigned k) {
    unsigned b = (k & 0x80000000u) ? (k & 0x7FFFFFFFu) : ~k;
    return __uint_as_float(b);
}

__global__ void gat_edge_score(const float* __restrict__ xl, const float* __restrict__ xr,
                               const float* __restrict__ att,
                               const int* __restrict__ src, const int* __restrict__ dst,
                               float* __restrict__ e_out, unsigned* __restrict__ m_enc,
                               int E, int H) {
    const int wave = (int)((blockIdx.x * (unsigned)blockDim.x + threadIdx.x) >> 5);
    const int lane = threadIdx.x & 31;
    if (wave >= E) return;
    if (lane == 0 && wave + 2048 < E) {            // stream-ahead on the index arrays
        __builtin_prefetch(src + wave + 2048, 0, 1);
        __builtin_prefetch(dst + wave + 2048, 0, 1);
    }
    const int s = src[wave];
    const int d = dst[wave];
    const float4* __restrict__ pl = (const float4*)(xl + (size_t)s * H);
    const float4* __restrict__ pr = (const float4*)(xr + (size_t)d * H);
    const float4* __restrict__ pa = (const float4*)att;
    float acc = 0.f;
    for (int f4 = lane; f4 * 4 < H; f4 += 32) {
        const float4 a = pl[f4];
        const float4 b = pr[f4];
        const float4 t = pa[f4];
        float v0 = a.x + b.x, v1 = a.y + b.y, v2 = a.z + b.z, v3 = a.w + b.w;
        v0 = (v0 > 0.f) ? v0 : v0 * NEG_SLOPE;
        v1 = (v1 > 0.f) ? v1 : v1 * NEG_SLOPE;
        v2 = (v2 > 0.f) ? v2 : v2 * NEG_SLOPE;
        v3 = (v3 > 0.f) ? v3 : v3 * NEG_SLOPE;
        acc += v0 * t.x + v1 * t.y + v2 * t.z + v3 * t.w;
    }
#pragma unroll
    for (int off = 16; off > 0; off >>= 1) acc += __shfl_xor(acc, off, 32);
    if (lane == 0) {
        e_out[wave] = acc;
        atomicMax(m_enc + d, enc_f32(acc));
    }
}

// ---------------------------------------------------------------------------
// Softmax numerator: ex = exp(e - m[dst]) (in place) ; s[dst] += ex
// ---------------------------------------------------------------------------
__global__ void gat_edge_softmax_num(const int* __restrict__ dst,
                                     const unsigned* __restrict__ m_enc,
                                     float* __restrict__ e_inout,
                                     float* __restrict__ s_sum, int E) {
    const int i = (int)(blockIdx.x * (unsigned)blockDim.x + threadIdx.x);
    if (i >= E) return;
    const int d = dst[i];
    const float m = dec_f32(m_enc[d]);
    const float ex = __expf(e_inout[i] - m);
    e_inout[i] = ex;
    atomicAdd(s_sum + d, ex);
}

// ---------------------------------------------------------------------------
// Weighted aggregation: agg[dst] += (ex/s[dst]) * xl[src]  (one wave per edge)
// ---------------------------------------------------------------------------
__global__ void gat_edge_aggregate(const float* __restrict__ xl,
                                   const int* __restrict__ src, const int* __restrict__ dst,
                                   const float* __restrict__ ex, const float* __restrict__ s_sum,
                                   float* __restrict__ agg, int E, int H) {
    const int wave = (int)((blockIdx.x * (unsigned)blockDim.x + threadIdx.x) >> 5);
    const int lane = threadIdx.x & 31;
    if (wave >= E) return;
    const int s = src[wave];
    const int d = dst[wave];
    const float alpha = ex[wave] / s_sum[d];
    const float4* __restrict__ pl = (const float4*)(xl + (size_t)s * H);
    float* __restrict__ pa = agg + (size_t)d * H;
    for (int f4 = lane; f4 * 4 < H; f4 += 32) {
        const float4 v = pl[f4];
        atomicAdd(pa + f4 * 4 + 0, alpha * v.x);
        atomicAdd(pa + f4 * 4 + 1, alpha * v.y);
        atomicAdd(pa + f4 * 4 + 2, alpha * v.z);
        atomicAdd(pa + f4 * 4 + 3, alpha * v.w);
    }
}

// ---------------------------------------------------------------------------
// out = relu(agg + bias + skip)
// ---------------------------------------------------------------------------
__global__ void gat_bias_skip_relu(const float* __restrict__ agg, const float* __restrict__ bias,
                                   const float* __restrict__ skip, float* __restrict__ out,
                                   int M, int H) {
    const size_t i = (size_t)blockIdx.x * blockDim.x + threadIdx.x;
    const size_t total = (size_t)M * H;
    if (i >= total) return;
    const float v = agg[i] + bias[(int)(i % H)] + skip[i];
    out[i] = (v > 0.f) ? v : 0.f;
}

// ---------------------------------------------------------------------------
extern "C" void kernel_launch(void* const* d_in, const int* in_sizes, int n_in,
                              void* d_out, int out_size, void* d_ws, size_t ws_size,
                              hipStream_t stream) {
    const int D_IN = 128, HID = 128, OUT = 64;
    const int Nn = in_sizes[0] / D_IN;     // 50000
    const int E  = in_sizes[1] / 2;        // 800000

    const float* x    = (const float*)d_in[0];
    const int*   ei   = (const int*)d_in[1];
    const int*   src  = ei;
    const int*   dst  = ei + E;
    const float* Wl1  = (const float*)d_in[2];
    const float* bl1  = (const float*)d_in[3];
    const float* Wr1  = (const float*)d_in[4];
    const float* br1  = (const float*)d_in[5];
    const float* att1 = (const float*)d_in[6];
    const float* bias1= (const float*)d_in[7];
    const float* Ws1  = (const float*)d_in[8];
    const float* bs1  = (const float*)d_in[9];
    const float* Wl2  = (const float*)d_in[10];
    const float* bl2  = (const float*)d_in[11];
    const float* Wr2  = (const float*)d_in[12];
    const float* br2  = (const float*)d_in[13];
    const float* att2 = (const float*)d_in[14];
    const float* bias2= (const float*)d_in[15];
    const float* Ws2  = (const float*)d_in[16];
    const float* bs2  = (const float*)d_in[17];

    // ---- workspace layout (floats) ----
    float* ws = (float*)d_ws;
    const size_t nh = (size_t)Nn * HID;
    float*    xl    = ws;                    // [N,HID] (reused as [N,OUT] in layer 2)
    float*    xr    = xl + nh;
    float*    skip  = xr + nh;
    float*    agg   = skip + nh;
    float*    h     = agg + nh;              // layer-1 output features [N,HID]
    float*    ebuf  = h + nh;                // [E] scores -> softmax numerators
    unsigned* menc  = (unsigned*)(ebuf + E); // [N] encoded segment max
    float*    ssum  = (float*)(menc + Nn);   // [N] segment sum

    const int mTiles     = (Nn + 15) / 16;          // 3125
    const int gemmBlocks = (mTiles + 7) / 8;        // 8 waves (M tiles) per block
    const int edgeBlocks = (E * 32 + 255) / 256;    // 1 wave per edge
    const int edgeTB     = (E + 255) / 256;

    // ================= Layer 1 (128 -> 128) =================
    gat_fused3_gemm<128, 128><<<gemmBlocks, 256, 128 * 128 * 2, stream>>>(
        x, Wl1, Wr1, Ws1, bl1, br1, bs1, xl, xr, skip, Nn);

    hipMemsetAsync(menc, 0, (size_t)Nn * sizeof(unsigned), stream);
    hipMemsetAsync(ssum, 0, (size_t)Nn * sizeof(float), stream);
    hipMemsetAsync(agg,  0, nh * sizeof(float), stream);

    gat_edge_score<<<edgeBlocks, 256, 0, stream>>>(xl, xr, att1, src, dst, ebuf, menc, E, HID);
    gat_edge_softmax_num<<<edgeTB, 256, 0, stream>>>(dst, menc, ebuf, ssum, E);
    gat_edge_aggregate<<<edgeBlocks, 256, 0, stream>>>(xl, src, dst, ebuf, ssum, agg, E, HID);

    gat_bias_skip_relu<<<(int)((nh + 255) / 256), 256, 0, stream>>>(agg, bias1, skip, h, Nn, HID);

    // ================= Layer 2 (128 -> 64) =================
    gat_fused3_gemm<128, 64><<<gemmBlocks, 256, 128 * 64 * 2, stream>>>(
        h, Wl2, Wr2, Ws2, bl2, br2, bs2, xl, xr, skip, Nn);

    hipMemsetAsync(menc, 0, (size_t)Nn * sizeof(unsigned), stream);
    hipMemsetAsync(ssum, 0, (size_t)Nn * sizeof(float), stream);
    hipMemsetAsync(agg,  0, (size_t)Nn * OUT * sizeof(float), stream);

    gat_edge_score<<<edgeBlocks, 256, 0, stream>>>(xl, xr, att2, src, dst, ebuf, menc, E, OUT);
    gat_edge_softmax_num<<<edgeTB, 256, 0, stream>>>(dst, menc, ebuf, ssum, E);
    gat_edge_aggregate<<<edgeBlocks, 256, 0, stream>>>(xl, src, dst, ebuf, ssum, agg, E, OUT);

    gat_bias_skip_relu<<<(int)(((size_t)Nn * OUT + 255) / 256), 256, 0, stream>>>(
        agg, bias2, skip, (float*)d_out, Nn, OUT);
}
